// HeteroGAT_63599875719678
// MI455X (gfx1250) — matrix operand order repaired and checked by reading, not compiled, not run on hardware
//
#include <hip/hip_runtime.h>
#include <hip/hip_bf16.h>

// ---------------------------------------------------------------------------
// HeteroGAT on MI455X (gfx1250): bf16 WMMA GEMMs with async global->LDS
// double-buffered staging + scatter softmax/aggregation.
// ---------------------------------------------------------------------------

typedef __attribute__((ext_vector_type(8)))  __bf16 v8bf;
typedef __attribute__((ext_vector_type(16))) __bf16 v16bf;
typedef __attribute__((ext_vector_type(8)))  float  v8f;

typedef __attribute__((address_space(3))) char lds_char_t;

#define NENT 100000
#define NLIT 50000
#define NE0  300000
#define NE1  200000
#define DHID 256
#define NHEAD 4

__device__ __forceinline__ __bf16 f2bf(float f) {
    unsigned u = __float_as_uint(f);
    u += 0x7FFFu + ((u >> 16) & 1u);            // round-to-nearest-even
    unsigned short hs = (unsigned short)(u >> 16);
    __bf16 r;
    __builtin_memcpy(&r, &hs, sizeof(r));
    return r;
}

// gfx1250 async memory->LDS copy (ASYNCcnt-tracked, bypasses VGPRs)
__device__ __forceinline__ void async_copy_b128(void* lds, const void* gp) {
    asm volatile("global_load_async_to_lds_b128 %0, %1, off"
                 :: "v"((lds_char_t*)lds), "v"(gp)
                 : "memory");
}
__device__ __forceinline__ void wait_asynccnt0() {
    asm volatile("s_wait_asynccnt 0" ::: "memory");
}

// ---------------------------------------------------------------------------
// GEMM: OUT[M,256] = act(A[M,K] @ W[K,256] + bias).
// A: bf16 row-major [M,K].  Wt: bf16 N-major [256,K] (pre-transposed).
// Block = 256 threads = 8 waves in a 2(M) x 4(N) grid; each wave owns a
// 2x2 set of 16x16 accumulators -> block tile 64(M) x 128(N).
// K-tiles of 32 staged into double-buffered LDS with async b128 copies;
// next tile's copies overlap the current tile's 4 WMMAs.
// A-fragment (16-bit A 16x32): lane<16 holds K {0..7,16..23}, lane>=16 holds
// K {8..15,24..31} of row M=lane&15. B-fragment (32x16): lane holds column
// N=lane&15, K {0..7,16..23} (lane<16) or {8..15,24..31} (lane>=16), as for A.
// ---------------------------------------------------------------------------
__global__ __launch_bounds__(256)
void gemm_bf16(const __bf16* __restrict__ A, const __bf16* __restrict__ Wt,
               const float* __restrict__ bias, float* __restrict__ OUT,
               int M, int K, int act)
{
    constexpr int LDA = 40;                       // padded stride (bf16), 80B rows
    __shared__ __bf16 sA[2][64 * LDA];            // 64 rows x 32 K
    __shared__ __bf16 sB[2][128 * LDA];           // 128 cols x 32 K

    const int tid  = threadIdx.x;
    const int lane = tid & 31;
    const int wave = tid >> 5;
    const int wm   = wave & 1;                    // wave tile row 0..1
    const int wn   = wave >> 1;                   // wave tile col 0..3
    const int row0 = blockIdx.x * 64;
    const int n0   = blockIdx.y * 128;

    // staging assignment: A 64x32 bf16 = 256 lanes x 16B; B 128x32 = 256 x 32B
    const int ar = tid >> 2;                      // 0..63
    const int ac = (tid & 3) << 3;                // 0,8,16,24 (bf16 units)
    const int bn = tid >> 1;                      // 0..127
    const int bs = (tid & 1) << 4;                // 0,16  (bf16 units)

    const int arr = (row0 + ar < M) ? (row0 + ar) : (M - 1);  // clamped row
    const __bf16* agp = A + (size_t)arr * K + ac;
    const __bf16* bgp = Wt + (size_t)(n0 + bn) * K + bs;

    v8f acc00 = {}, acc01 = {}, acc10 = {}, acc11 = {};

    auto stage = [&](int p, int k0) {
        async_copy_b128(&sA[p][ar * LDA + ac], agp + k0);
        async_copy_b128(&sB[p][bn * LDA + bs], bgp + k0);
        async_copy_b128(&sB[p][bn * LDA + bs + 8], bgp + k0 + 8);
    };

    stage(0, 0);
    int p = 0;
    for (int k0 = 0; k0 < K; k0 += 32, p ^= 1) {
        wait_asynccnt0();                         // my copies for buffer p done
        __syncthreads();                          // everyone's copies done
        if (k0 + 32 < K) stage(p ^ 1, k0 + 32);   // overlap next tile with WMMA

        const int akb  = (lane >> 4) * 8;
        const int bkb  = (lane >> 4) * 8;
        const int amr0 = (wm * 32 + (lane & 15)) * LDA;
        const int amr1 = amr0 + 16 * LDA;
        const int bcr0 = (wn * 32 + (lane & 15)) * LDA;
        const int bcr1 = bcr0 + 16 * LDA;

        v8bf a0lo = *(const v8bf*)&sA[p][amr0 + akb];
        v8bf a0hi = *(const v8bf*)&sA[p][amr0 + akb + 16];
        v8bf a1lo = *(const v8bf*)&sA[p][amr1 + akb];
        v8bf a1hi = *(const v8bf*)&sA[p][amr1 + akb + 16];
        v8bf q0lo = *(const v8bf*)&sB[p][bcr0 + bkb];
        v8bf q0hi = *(const v8bf*)&sB[p][bcr0 + bkb + 16];
        v8bf q1lo = *(const v8bf*)&sB[p][bcr1 + bkb];
        v8bf q1hi = *(const v8bf*)&sB[p][bcr1 + bkb + 16];

        v16bf af0, af1, bf0, bf1;
        #pragma unroll
        for (int j = 0; j < 8; ++j) {
            af0[j] = a0lo[j]; af0[j + 8] = a0hi[j];
            af1[j] = a1lo[j]; af1[j + 8] = a1hi[j];
            bf0[j] = q0lo[j]; bf0[j + 8] = q0hi[j];
            bf1[j] = q1lo[j]; bf1[j + 8] = q1hi[j];
        }
        acc00 = __builtin_amdgcn_wmma_f32_16x16x32_bf16(false, af0, false, bf0, (short)0, acc00, false, false);
        acc01 = __builtin_amdgcn_wmma_f32_16x16x32_bf16(false, af0, false, bf1, (short)0, acc01, false, false);
        acc10 = __builtin_amdgcn_wmma_f32_16x16x32_bf16(false, af1, false, bf0, (short)0, acc10, false, false);
        acc11 = __builtin_amdgcn_wmma_f32_16x16x32_bf16(false, af1, false, bf1, (short)0, acc11, false, false);
        __syncthreads();                          // readers done before rewrite
    }

    // --- epilogue: bias + activation ---
    // C layout: VGPR r -> M = r + 8*(lane>=16), N = lane&15
    const int cin  = lane & 15;
    const int rofs = (lane >> 4) << 3;
    const int col0 = n0 + wn * 32 + cin;
    const int col1 = col0 + 16;
    const float bb0 = bias[col0];
    const float bb1 = bias[col1];
    #pragma unroll
    for (int r = 0; r < 8; ++r) {
        int rowa = row0 + wm * 32 + r + rofs;     // tiles (0,*)
        int rowb = rowa + 16;                     // tiles (1,*)
        if (rowa < M) {
            float v0 = acc00[r] + bb0, v1 = acc01[r] + bb1;
            if (act) { v0 = v0 > 0.f ? v0 : 0.f;  v1 = v1 > 0.f ? v1 : 0.f; }
            OUT[(size_t)rowa * DHID + col0] = v0;
            OUT[(size_t)rowa * DHID + col1] = v1;
        }
        if (rowb < M) {
            float v0 = acc10[r] + bb0, v1 = acc11[r] + bb1;
            if (act) { v0 = v0 > 0.f ? v0 : 0.f;  v1 = v1 > 0.f ? v1 : 0.f; }
            OUT[(size_t)rowb * DHID + col0] = v0;
            OUT[(size_t)rowb * DHID + col1] = v1;
        }
    }
}

// f32 -> bf16 elementwise
__global__ __launch_bounds__(256)
void cvt_f32_bf16(const float* __restrict__ in, __bf16* __restrict__ out, size_t n)
{
    size_t i = (size_t)blockIdx.x * 256 + threadIdx.x;
    if (i < n) out[i] = f2bf(in[i]);
}

// W[K,256] f32 -> Wt[256,K] bf16 (N-major, K contiguous)
__global__ __launch_bounds__(256)
void transpose_cvt(const float* __restrict__ W, __bf16* __restrict__ Wt, int K)
{
    int i = blockIdx.x * 256 + threadIdx.x;
    if (i >= K * 256) return;
    int k = i >> 8, n = i & 255;
    Wt[(size_t)n * K + k] = f2bf(W[(size_t)k * 256 + n]);
}

// ---------------------------------------------------------------------------
// s[n,h] = sum_d m[n, h*64+d] * avec[h*64+d]  (one wave per (n,h))
// ---------------------------------------------------------------------------
__global__ __launch_bounds__(256)
void head_scores(const float* __restrict__ m, const float* __restrict__ avec,
                 float* __restrict__ s, int n)
{
    int id   = blockIdx.x * 8 + (threadIdx.x >> 5);
    int lane = threadIdx.x & 31;
    if (id >= n * NHEAD) return;
    int node = id >> 2, h = id & 3;
    const float* mp = m + (size_t)node * DHID + h * 64;
    const float* ap = avec + h * 64;
    float p = mp[lane] * ap[lane] + mp[lane + 32] * ap[lane + 32];
    #pragma unroll
    for (int off = 16; off; off >>= 1) p += __shfl_xor(p, off, 32);
    if (lane == 0) s[id] = p;
}

__device__ __forceinline__ void atomicMaxF(float* addr, float val) {
    unsigned* ua = (unsigned*)addr;
    unsigned old = *ua;
    while (true) {
        if (__uint_as_float(old) >= val) break;
        unsigned assumed = old;
        old = atomicCAS(ua, assumed, __float_as_uint(val));
        if (old == assumed) break;
    }
}

// logit = leaky_relu(s_src[src] + s_dst[dst] + s_e); running segment max.
__global__ __launch_bounds__(256)
void edge_logit(const int* __restrict__ src, const int* __restrict__ dst,
                const float* __restrict__ ssrc, const float* __restrict__ sdst,
                const float* __restrict__ se, float* __restrict__ logit,
                float* __restrict__ maxv, int E)
{
    int i = blockIdx.x * 256 + threadIdx.x;       // i = e*H + h
    if (i >= E * NHEAD) return;
    int e = i >> 2, h = i & 3;
    int s = src[e], d = dst[e];
    float x = ssrc[s * NHEAD + h] + sdst[d * NHEAD + h] + se[i];
    x = x > 0.f ? x : 0.2f * x;                   // leaky relu, slope 0.2
    logit[i] = x;
    atomicMaxF(&maxv[d * NHEAD + h], x);
}

// ex = exp(logit - max[dst]); segment sum via atomics; overwrite logit with ex.
__global__ __launch_bounds__(256)
void edge_expsum(const int* __restrict__ dst, float* __restrict__ logit,
                 const float* __restrict__ maxv, float* __restrict__ sumv, int E)
{
    int i = blockIdx.x * 256 + threadIdx.x;
    if (i >= E * NHEAD) return;
    int e = i >> 2, h = i & 3;
    int d = dst[e];
    float ex = __expf(logit[i] - maxv[d * NHEAD + h]);
    logit[i] = ex;
    atomicAdd(&sumv[d * NHEAD + h], ex);
}

// agg[dst, :] += alpha * (m_src[src, :] + ep[e, :])   (one block per edge)
__global__ __launch_bounds__(256)
void edge_aggregate(const int* __restrict__ src, const int* __restrict__ dst,
                    const float* __restrict__ ex, const float* __restrict__ sumv,
                    const float* __restrict__ msrc, const float* __restrict__ ep,
                    float* __restrict__ agg, int E)
{
    int e = blockIdx.x;
    int d = threadIdx.x;                          // 0..255
    int h = d >> 6;
    int s = src[e], t = dst[e];
    float alpha = ex[(size_t)e * NHEAD + h] / (sumv[t * NHEAD + h] + 1e-16f);
    float v = alpha * (msrc[(size_t)s * DHID + d] + ep[(size_t)e * DHID + d]);
    atomicAdd(&agg[(size_t)t * DHID + d], v);
}

__global__ __launch_bounds__(256)
void residual_elu(const float* __restrict__ agg, float* __restrict__ h, size_t n)
{
    size_t i = (size_t)blockIdx.x * 256 + threadIdx.x;
    if (i >= n) return;
    float x = agg[i] + h[i];
    h[i] = x > 0.f ? x : (__expf(x) - 1.f);
}

__global__ __launch_bounds__(256)
void fillk(float* __restrict__ p, float v, size_t n)
{
    size_t i = (size_t)blockIdx.x * 256 + threadIdx.x;
    if (i < n) p[i] = v;
}

// ---------------------------------------------------------------------------
// Host-side orchestration
// ---------------------------------------------------------------------------
static inline void launch_gemm(const __bf16* A, const __bf16* Wt, const float* b,
                               float* OUT, int M, int K, int act, hipStream_t s)
{
    dim3 grid((M + 63) / 64, 2);
    gemm_bf16<<<grid, 256, 0, s>>>(A, Wt, b, OUT, M, K, act);
}

static inline void launch_fill(float* p, float v, size_t n, hipStream_t s)
{
    fillk<<<(unsigned)((n + 255) / 256), 256, 0, s>>>(p, v, n);
}

static inline void launch_cvt(const float* in, __bf16* out, size_t n, hipStream_t s)
{
    cvt_f32_bf16<<<(unsigned)((n + 255) / 256), 256, 0, s>>>(in, out, n);
}

static void etype_conv(hipStream_t stream,
                       const __bf16* hb_src, const __bf16* hb_dst,
                       const int* src, const int* dst, int E,
                       const __bf16* eattr_bf,
                       const __bf16* wt, const __bf16* wet, const float* zbias,
                       const float* asrc, const float* adst, const float* ae,
                       int n_src, int n_dst,
                       float* m_src, float* m_dst,          // m_dst may == m_src
                       float* ep, float* ssrc, float* sdst, float* se,
                       float* maxv, float* sumv, float* logit, float* agg)
{
    launch_gemm(hb_src, wt, zbias, m_src, n_src, DHID, 0, stream);
    if (m_dst != m_src)
        launch_gemm(hb_dst, wt, zbias, m_dst, n_dst, DHID, 0, stream);
    launch_gemm(eattr_bf, wet, zbias, ep, E, DHID, 0, stream);

    head_scores<<<(n_src * NHEAD + 7) / 8, 256, 0, stream>>>(m_src, asrc, ssrc, n_src);
    head_scores<<<(n_dst * NHEAD + 7) / 8, 256, 0, stream>>>(m_dst, adst, sdst, n_dst);
    head_scores<<<(E * NHEAD + 7) / 8, 256, 0, stream>>>(ep, ae, se, E);

    launch_fill(maxv, -3.0e38f, (size_t)n_dst * NHEAD, stream);
    launch_fill(sumv, 0.f, (size_t)n_dst * NHEAD, stream);
    launch_fill(agg, 0.f, (size_t)n_dst * DHID, stream);

    int eb = (E * NHEAD + 255) / 256;
    edge_logit<<<eb, 256, 0, stream>>>(src, dst, ssrc, sdst, se, logit, maxv, E);
    edge_expsum<<<eb, 256, 0, stream>>>(dst, logit, maxv, sumv, E);
    edge_aggregate<<<E, 256, 0, stream>>>(src, dst, logit, sumv, m_src, ep, agg, E);
}

extern "C" void kernel_launch(void* const* d_in, const int* in_sizes, int n_in,
                              void* d_out, int out_size, void* d_ws, size_t ws_size,
                              hipStream_t stream)
{
    const float* x_ent   = (const float*)d_in[0];
    const float* x_lit   = (const float*)d_in[1];
    const float* ea0     = (const float*)d_in[2];
    const float* ea1     = (const float*)d_in[3];
    const float* w_in_e  = (const float*)d_in[4];
    const float* b_in_e  = (const float*)d_in[5];
    const float* w_in_l  = (const float*)d_in[6];
    const float* b_in_l  = (const float*)d_in[7];
    const float* w_e0    = (const float*)d_in[8];
    const float* we_e0   = (const float*)d_in[9];
    const float* asrc_e0 = (const float*)d_in[10];
    const float* adst_e0 = (const float*)d_in[11];
    const float* ae_e0   = (const float*)d_in[12];
    const float* w_e1    = (const float*)d_in[13];
    const float* we_e1   = (const float*)d_in[14];
    const float* asrc_e1 = (const float*)d_in[15];
    const float* adst_e1 = (const float*)d_in[16];
    const float* ae_e1   = (const float*)d_in[17];
    const int*   ei0     = (const int*)d_in[18];
    const int*   ei1     = (const int*)d_in[19];

    const int* src0 = ei0;        const int* dst0 = ei0 + NE0;
    const int* src1 = ei1;        const int* dst1 = ei1 + NE1;

    float* h_ent = (float*)d_out;
    float* h_lit = h_ent + (size_t)NENT * DHID;

    // ---- workspace carve-out (f32 region, then bf16 region) ----
    float* ws = (float*)d_ws;
    size_t o = 0;
    float* m_sh  = ws + o; o += (size_t)NENT * DHID;   // m0 / m1_src
    float* m1d   = ws + o; o += (size_t)NLIT * DHID;   // m1_dst
    float* ep    = ws + o; o += (size_t)NE0 * DHID;    // edge proj (NE0 >= NE1)
    float* agg_e = ws + o; o += (size_t)NENT * DHID;
    float* agg_l = ws + o; o += (size_t)NLIT * DHID;
    float* ssrc  = ws + o; o += (size_t)NENT * NHEAD;
    float* sdst  = ws + o; o += (size_t)NENT * NHEAD;
    float* se    = ws + o; o += (size_t)NE0 * NHEAD;
    float* maxv  = ws + o; o += (size_t)NENT * NHEAD;
    float* sumv  = ws + o; o += (size_t)NENT * NHEAD;
    float* logit = ws + o; o += (size_t)NE0 * NHEAD;
    float* zbias = ws + o; o += 256;                   // always-valid zero bias

    __bf16* bws = (__bf16*)(ws + o);
    size_t ob = 0;
    // xbuf (NENT*768 = 76.8M) exactly matches ea0 bf16 (NE0*256) -> alias
    __bf16* xbuf   = bws + ob; ob += (size_t)NENT * 768;   // also ea0_bf
    __bf16* ea0bf  = xbuf;
    __bf16* ea1bf  = bws + ob; ob += (size_t)NE1 * DHID;
    __bf16* hbfe   = bws + ob; ob += (size_t)NENT * DHID;
    __bf16* hbfl   = bws + ob; ob += (size_t)NLIT * DHID;
    __bf16* wt_in_e = bws + ob; ob += (size_t)768 * DHID;
    __bf16* wt_in_l = bws + ob; ob += (size_t)768 * DHID;
    __bf16* wt_e0   = bws + ob; ob += 2 * (size_t)DHID * DHID;
    __bf16* wte_e0  = bws + ob; ob += 2 * (size_t)DHID * DHID;
    __bf16* wt_e1   = bws + ob; ob += 2 * (size_t)DHID * DHID;
    __bf16* wte_e1  = bws + ob; ob += 2 * (size_t)DHID * DHID;
    (void)ob; (void)ws_size; (void)in_sizes; (void)n_in; (void)out_size;

    launch_fill(zbias, 0.f, 256, stream);

    // ---- pre-transpose + convert all weights to bf16 N-major ----
    const size_t PL = (size_t)DHID * DHID;   // per-layer weight stride
    const size_t PA = (size_t)NHEAD * 64;    // per-layer attn-vector stride
    transpose_cvt<<<(768 * 256) / 256, 256, 0, stream>>>(w_in_e, wt_in_e, 768);
    transpose_cvt<<<(768 * 256) / 256, 256, 0, stream>>>(w_in_l, wt_in_l, 768);
    for (int l = 0; l < 2; ++l) {
        transpose_cvt<<<(256 * 256) / 256, 256, 0, stream>>>(w_e0 + l * PL,  wt_e0 + l * PL,  256);
        transpose_cvt<<<(256 * 256) / 256, 256, 0, stream>>>(we_e0 + l * PL, wte_e0 + l * PL, 256);
        transpose_cvt<<<(256 * 256) / 256, 256, 0, stream>>>(w_e1 + l * PL,  wt_e1 + l * PL,  256);
        transpose_cvt<<<(256 * 256) / 256, 256, 0, stream>>>(we_e1 + l * PL, wte_e1 + l * PL, 256);
    }

    // ---- input projections + ReLU (x converted to bf16 through xbuf) ----
    launch_cvt(x_ent, xbuf, (size_t)NENT * 768, stream);
    launch_gemm(xbuf, wt_in_e, b_in_e, h_ent, NENT, 768, 1, stream);
    launch_cvt(x_lit, xbuf, (size_t)NLIT * 768, stream);
    launch_gemm(xbuf, wt_in_l, b_in_l, h_lit, NLIT, 768, 1, stream);

    // ---- edge attributes converted once (reused by both layers) ----
    launch_cvt(ea0, ea0bf, (size_t)NE0 * DHID, stream);   // overwrites xbuf (done)
    launch_cvt(ea1, ea1bf, (size_t)NE1 * DHID, stream);

    for (int l = 0; l < 2; ++l) {
        // convert current h to bf16 for this layer's GEMMs
        launch_cvt(h_ent, hbfe, (size_t)NENT * DHID, stream);
        launch_cvt(h_lit, hbfl, (size_t)NLIT * DHID, stream);

        // edge type 0: entity -> entity
        etype_conv(stream, hbfe, hbfe, src0, dst0, NE0, ea0bf,
                   wt_e0 + l * PL, wte_e0 + l * PL, zbias,
                   asrc_e0 + l * PA, adst_e0 + l * PA, ae_e0 + l * PA,
                   NENT, NENT, m_sh, m_sh, ep, ssrc, sdst, se,
                   maxv, sumv, logit, agg_e);
        // edge type 1: entity -> literal
        etype_conv(stream, hbfe, hbfl, src1, dst1, NE1, ea1bf,
                   wt_e1 + l * PL, wte_e1 + l * PL, zbias,
                   asrc_e1 + l * PA, adst_e1 + l * PA, ae_e1 + l * PA,
                   NENT, NLIT, m_sh, m1d, ep, ssrc, sdst, se,
                   maxv, sumv, logit, agg_l);
        // residual + ELU (after both convs, as in reference)
        residual_elu<<<(unsigned)(((size_t)NENT * DHID + 255) / 256), 256, 0, stream>>>(
            agg_e, h_ent, (size_t)NENT * DHID);
        residual_elu<<<(unsigned)(((size_t)NLIT * DHID + 255) / 256), 256, 0, stream>>>(
            agg_l, h_lit, (size_t)NLIT * DHID);
    }
}
